// SimpleGatedUnit1_13795434954909
// MI455X (gfx1250) — compile-verified, matrix operand order checked by reading, not run on hardware
//
#include <hip/hip_runtime.h>
#include <hip/hip_bf16.h>

// -----------------------------------------------------------------------------
// SimpleGatedUnit on MI455X (gfx1250, wave32, WMMA bf16)
//
// Phase 1: transpose+cast the 6 weight matrices (W*, U*) to bf16, N-major so
//          each WMMA B-fragment is a contiguous 32B load per lane.
// Phase 2: big GEMM  xx/x_in/x_out = Xt@W* + b*  with v_wmma_f32_16x16x32_bf16,
//          A fragment built on the fly from fp32 X. Outputs stored bf16:
//          3 x 32MB = 96MB -> the whole scan working set stays in the 192MB L2.
// Phase 3: persistent scan kernel. 4 workgroups (16 batch rows each, the
//          recurrence is independent across batch), 32 waves per WG (one 16-col
//          N tile each). h slice lives in LDS as bf16 (A operand), fp32 state
//          in registers. Two __syncthreads per step. Gate-data loads are
//          software-pipelined ahead of the WMMA loops and the next step's
//          slices are warmed with global_prefetch_b8.
// -----------------------------------------------------------------------------

typedef __attribute__((ext_vector_type(16))) __bf16 v16bf;
typedef __attribute__((ext_vector_type(8)))  __bf16 v8bf;
typedef __attribute__((ext_vector_type(8)))  float  v8f;
typedef __attribute__((ext_vector_type(4)))  float  v4f;

#define T_STEPS   512
#define BATCH     64
#define DIM       512
#define LDS_PITCH 520   // 512 + 8 bf16 pad to break LDS bank alignment

__device__ __forceinline__ float hsig(float x) {
    return fminf(fmaxf(0.2f * x + 0.5f, 0.0f), 1.0f);
}
__device__ __forceinline__ float softplus_f(float x) {
    // numerically stable log(1+exp(x))
    return fmaxf(x, 0.0f) + log1pf(expf(-fabsf(x)));
}

// Load a 16-element bf16 fragment from LDS as two 16B v8bf loads
// (row pitch 1040B is 16B- but not 32B-aligned).
__device__ __forceinline__ v16bf frag_lds(const __bf16* p) {
    v8bf lo = *(const v8bf*)p;
    v8bf hi = *(const v8bf*)(p + 8);
    v16bf r;
#pragma unroll
    for (int i = 0; i < 8; ++i) { r[i] = lo[i]; r[i + 8] = hi[i]; }
    return r;
}

// ---------------------------------------------------------------------------
// Kernel 1: 512x512 fp32 (k,n) -> bf16 (n,k)   (weights transpose + cast)
// ---------------------------------------------------------------------------
__global__ void transpose_bf16_kernel(const float* __restrict__ src,
                                      __bf16* __restrict__ dst) {
    int idx = blockIdx.x * blockDim.x + threadIdx.x;   // 0 .. 262143
    int n = idx >> 9;
    int k = idx & 511;
    dst[(size_t)n * DIM + k] = (__bf16)src[(size_t)k * DIM + n];
}

// ---------------------------------------------------------------------------
// Kernel 2: xx/x_in/x_out = Xt @ W* + b*   (M = T*B = 32768, K = N = 512)
// One wave per 16x16 output tile; A fragment reused for all three weights.
// A layout: lanes 0-15 -> M=lane, K chunk first half; lanes 16-31 -> second
// K half (same contiguous-per-lane K ordering used for B, so k-pairing is
// consistent). C layout: VGPR r, lane l -> M = r + (l>=16)*8, N = l&15.
// ---------------------------------------------------------------------------
__global__ void __launch_bounds__(256) input_gemm_kernel(
    const float*  __restrict__ X,
    const __bf16* __restrict__ Wt0, const __bf16* __restrict__ Wt1,
    const __bf16* __restrict__ Wt2,
    const float*  __restrict__ bias0, const float* __restrict__ bias1,
    const float*  __restrict__ bias2,
    __bf16* __restrict__ out0, __bf16* __restrict__ out1,
    __bf16* __restrict__ out2)
{
    int wid   = blockIdx.x * 8 + (threadIdx.x >> 5);
    int lane  = threadIdx.x & 31;
    int mtile = wid >> 5;          // 0..2047  (T*B/16)
    int ntile = wid & 31;          // 0..31    (D/16)
    int m0 = mtile << 4;           // 16 consecutive rows share the same t
    int t  = m0 >> 6;
    int b0 = m0 & 63;
    int hi = lane >> 4;
    int lm = lane & 15;
    int n  = (ntile << 4) + lm;

    // X is (B, T, I): row for this lane's A matrix row (m = lm)
    const float* arow = X + ((size_t)(b0 + lm) * T_STEPS + t) * DIM;

    v8f acc0 = {}; v8f acc1 = {}; v8f acc2 = {};
#pragma unroll
    for (int c = 0; c < 16; ++c) {
        int kb = c * 32 + hi * 16;
        v16bf a;
        const v4f* ap = (const v4f*)(arow + kb);
#pragma unroll
        for (int q = 0; q < 4; ++q) {
            v4f v = ap[q];
#pragma unroll
            for (int e = 0; e < 4; ++e) a[q * 4 + e] = (__bf16)v[e];
        }
        v16bf bw0 = *(const v16bf*)(Wt0 + (size_t)n * DIM + kb);
        v16bf bw1 = *(const v16bf*)(Wt1 + (size_t)n * DIM + kb);
        v16bf bw2 = *(const v16bf*)(Wt2 + (size_t)n * DIM + kb);
        acc0 = __builtin_amdgcn_wmma_f32_16x16x32_bf16(false, a, false, bw0,
                                                       (short)0, acc0, false, false);
        acc1 = __builtin_amdgcn_wmma_f32_16x16x32_bf16(false, a, false, bw1,
                                                       (short)0, acc1, false, false);
        acc2 = __builtin_amdgcn_wmma_f32_16x16x32_bf16(false, a, false, bw2,
                                                       (short)0, acc2, false, false);
    }
    float bb0 = bias0[n], bb1 = bias1[n], bb2 = bias2[n];
#pragma unroll
    for (int r = 0; r < 8; ++r) {
        int m = r + hi * 8;
        size_t o = ((size_t)(t * BATCH + b0 + m)) * DIM + n;   // (T,B,D) layout
        out0[o] = (__bf16)(acc0[r] + bb0);
        out1[o] = (__bf16)(acc1[r] + bb1);
        out2[o] = (__bf16)(acc2[r] + bb2);
    }
}

// ---------------------------------------------------------------------------
// Kernel 3: persistent recurrence. grid = 4 (batch slices of 16 rows),
// block = 1024 (32 waves; wave w owns output columns [16w, 16w+16)).
// Per step:
//   phase A: issue gate loads, then accz = h@U, acci = h@U_in (h from LDS
//            bf16); z, z_in gates; write g = z_in (*) h to g_lds; barrier.
//   phase B: issue gate loads, then acco = g@U_out; z_out = softplus(...);
//            h update in fp32 regs; write bf16 h back to h_lds; barrier.
// h_lds reads all happen before barrier 1; h_lds writes happen in phase B
// (which only reads g_lds) -> the two barriers fully order the buffers.
// ---------------------------------------------------------------------------
__global__ void __launch_bounds__(1024) scan_kernel(
    const __bf16* __restrict__ Ut,  const __bf16* __restrict__ Uti,
    const __bf16* __restrict__ Uto,
    const __bf16* __restrict__ xa,  const __bf16* __restrict__ xi,
    const __bf16* __restrict__ xo,
    const float*  __restrict__ bu_in, const float* __restrict__ bu_out,
    float* __restrict__ out)
{
    __shared__ __bf16 h_lds[16][LDS_PITCH];
    __shared__ __bf16 g_lds[16][LDS_PITCH];

    int wave = threadIdx.x >> 5;   // 0..31 == N tile
    int lane = threadIdx.x & 31;
    int hi   = lane >> 4;
    int lm   = lane & 15;
    int n    = wave * 16 + lm;
    int wgb0 = blockIdx.x * 16;    // this WG's batch-row base

    for (int i = threadIdx.x; i < 16 * LDS_PITCH; i += 1024)
        (&h_lds[0][0])[i] = (__bf16)0.0f;

    v8f h   = {};                  // fp32 state, rows m = r + hi*8, column n
    v8f z   = {};
    v8f zin = {};
    float bin  = bu_in[n];
    float bout = bu_out[n];
    const __bf16* uz_base = Ut  + (size_t)n * DIM;   // B operand rows (n-major)
    const __bf16* ui_base = Uti + (size_t)n * DIM;
    const __bf16* uo_base = Uto + (size_t)n * DIM;
    __syncthreads();

    for (int s = 0; s < T_STEPS - 1; ++s) {
        // ---- warm next-but-one step's gate slices (global_prefetch_b8) --
        if (s + 2 < T_STEPS) {
            size_t pofs = ((size_t)(s + 2) * BATCH + wgb0 + lm) * DIM + (wave << 4);
            __builtin_prefetch((const void*)((hi ? xi : xa) + pofs), 0, 3);
            if (!hi) __builtin_prefetch((const void*)(xo + pofs), 0, 3);
        }

        // ---- phase A: issue this step's gate loads first so they overlap
        //      the 16-chunk WMMA loop below ------------------------------
        const __bf16* pa1 = xa + ((size_t)(s + 1) * BATCH + wgb0) * DIM;
        const __bf16* pa0 = xa + ((size_t)s       * BATCH + wgb0) * DIM;
        const __bf16* pi1 = xi + ((size_t)(s + 1) * BATCH + wgb0) * DIM;
        const __bf16* pi0 = xi + ((size_t)s       * BATCH + wgb0) * DIM;
        float ga1[8], ga0[8], gi1[8], gi0[8];
#pragma unroll
        for (int r = 0; r < 8; ++r) {
            size_t o = (size_t)(r + hi * 8) * DIM + n;
            ga1[r] = (float)pa1[o];
            ga0[r] = (float)pa0[o];
            gi1[r] = (float)pi1[o];
            gi0[r] = (float)pi0[o];
        }

        v8f accz = {}; v8f acci = {};
#pragma unroll 4
        for (int c = 0; c < 16; ++c) {
            int kb = c * 32 + hi * 16;
            v16bf a  = frag_lds(&h_lds[lm][kb]);
            v16bf bz = *(const v16bf*)(uz_base + kb);
            v16bf bi = *(const v16bf*)(ui_base + kb);
            accz = __builtin_amdgcn_wmma_f32_16x16x32_bf16(false, a, false, bz,
                                                           (short)0, accz, false, false);
            acci = __builtin_amdgcn_wmma_f32_16x16x32_bf16(false, a, false, bi,
                                                           (short)0, acci, false, false);
        }
#pragma unroll
        for (int r = 0; r < 8; ++r) {
            int m = r + hi * 8;
            z[r]   = hsig(ga1[r] + ga0[r] + accz[r]);
            zin[r] = hsig(gi1[r] + gi0[r] + acci[r] + bin);
            g_lds[m][n] = (__bf16)(zin[r] * h[r]);     // g = z_in (*) h
        }
        __syncthreads();

        // ---- phase B: gate loads first, then (z_in*h) @ U_out -----------
        const __bf16* po1 = xo + ((size_t)(s + 1) * BATCH + wgb0) * DIM;
        const __bf16* po0 = xo + ((size_t)s       * BATCH + wgb0) * DIM;
        float go1[8], go0[8];
#pragma unroll
        for (int r = 0; r < 8; ++r) {
            size_t o = (size_t)(r + hi * 8) * DIM + n;
            go1[r] = (float)po1[o];
            go0[r] = (float)po0[o];
        }

        v8f acco = {};
#pragma unroll 4
        for (int c = 0; c < 16; ++c) {
            int kb = c * 32 + hi * 16;
            v16bf a  = frag_lds(&g_lds[lm][kb]);
            v16bf bo = *(const v16bf*)(uo_base + kb);
            acco = __builtin_amdgcn_wmma_f32_16x16x32_bf16(false, a, false, bo,
                                                           (short)0, acco, false, false);
        }
#pragma unroll
        for (int r = 0; r < 8; ++r) {
            int m = r + hi * 8;
            float zo = softplus_f(go1[r] + go0[r] + acco[r] + bout);
            h[r] = (1.0f - z[r]) * h[r] + z[r] * zo;
            h_lds[m][n] = (__bf16)h[r];
        }
        __syncthreads();
    }

    // h_last -> d_out, shape (B, D) fp32
#pragma unroll
    for (int r = 0; r < 8; ++r) {
        int m = r + hi * 8;
        out[(size_t)(wgb0 + m) * DIM + n] = h[r];
    }
}

// ---------------------------------------------------------------------------
// Host side
// ---------------------------------------------------------------------------
extern "C" void kernel_launch(void* const* d_in, const int* in_sizes, int n_in,
                              void* d_out, int out_size, void* d_ws, size_t ws_size,
                              hipStream_t stream) {
    (void)in_sizes; (void)n_in; (void)out_size; (void)ws_size;
    const float* X      = (const float*)d_in[0];
    const float* W      = (const float*)d_in[1];
    const float* b      = (const float*)d_in[2];
    const float* W_in   = (const float*)d_in[3];
    const float* b_in   = (const float*)d_in[4];
    const float* W_out  = (const float*)d_in[5];
    const float* b_out  = (const float*)d_in[6];
    const float* U      = (const float*)d_in[7];
    const float* U_in   = (const float*)d_in[8];
    const float* U_out  = (const float*)d_in[9];
    const float* bu_in  = (const float*)d_in[10];
    const float* bu_out = (const float*)d_in[11];

    // Workspace layout (bytes):
    //   6 x 512KB bf16 transposed weights = 3MB, then 3 x 32MB bf16 activations.
    // Total ~99MB; the activation streams fit in the 192MB L2 for the scan.
    char* ws = (char*)d_ws;
    const size_t WMAT = (size_t)DIM * DIM;           // 262144 elements
    __bf16* Wt0 = (__bf16*)(ws);
    __bf16* Wt1 = Wt0 + WMAT;
    __bf16* Wt2 = Wt1 + WMAT;
    __bf16* Ut  = Wt2 + WMAT;
    __bf16* Uti = Ut  + WMAT;
    __bf16* Uto = Uti + WMAT;
    const size_t ACT = (size_t)T_STEPS * BATCH * DIM;   // 16,777,216 elements
    __bf16* xa = Uto + WMAT;
    __bf16* xi = xa + ACT;
    __bf16* xo = xi + ACT;

    // Phase 1: weight transpose + bf16 cast (6 x 512x512)
    dim3 tb(256), tg(WMAT / 256);
    transpose_bf16_kernel<<<tg, tb, 0, stream>>>(W,     Wt0);
    transpose_bf16_kernel<<<tg, tb, 0, stream>>>(W_in,  Wt1);
    transpose_bf16_kernel<<<tg, tb, 0, stream>>>(W_out, Wt2);
    transpose_bf16_kernel<<<tg, tb, 0, stream>>>(U,     Ut);
    transpose_bf16_kernel<<<tg, tb, 0, stream>>>(U_in,  Uti);
    transpose_bf16_kernel<<<tg, tb, 0, stream>>>(U_out, Uto);

    // Phase 2: input projections. 2048 M-tiles x 32 N-tiles, 8 waves/block.
    input_gemm_kernel<<<8192, 256, 0, stream>>>(X, Wt0, Wt1, Wt2,
                                                b, b_in, b_out,
                                                xa, xi, xo);

    // Phase 3: recurrence. 4 batch slices x 32 waves.
    scan_kernel<<<4, 1024, 0, stream>>>(Ut, Uti, Uto, xa, xi, xo,
                                        bu_in, bu_out, (float*)d_out);
}
